// PointNetPP_DownModule_90185723281828
// MI455X (gfx1250) — matrix operand order repaired and checked by reading, not compiled, not run on hardware
//
#include <hip/hip_runtime.h>

typedef _Float16 f16;
typedef __attribute__((ext_vector_type(16))) _Float16 v16h;
typedef __attribute__((ext_vector_type(8)))  float    v8f;

#define BZ      8
#define NPTS    4096
#define CFEAT   64
#define NSAMP   1024
#define KNN     64
#define RADIUS  0.2f
#define BN_EPS  1e-5f
#define NEGV    -1e8f
#define KP0     96          // 67 input channels padded to 3*32 for K-steps

// folded-weight storage (f16, swizzled into B-fragment tile layout)
#define W0_ELEMS (96*64)
#define W1_ELEMS (64*64)
#define W2_ELEMS (64*128)

// workspace byte offsets (~4.1 MB total)
#define WH_OFF   0
#define BIAS_OFF 36864
#define FPS_OFF  37888
#define TKI_OFF  70656
#define TKD_OFF  2167808

// ---------------- WMMA fragment helpers (CDNA5 wave32 layouts) ----------------
// 16-bit A 16x32: lanes 0-15 row M=lane, elems i -> K = i + 8*half + (i&8)
// (two contiguous 16B runs per lane -> ds_load_b128 x2)
__device__ __forceinline__ v16h ldsA_frag(const f16* base, int lda, int lane) {
  int half = (lane >> 4) & 1;
  int row  = lane & 15;
  v16h a;
#pragma unroll
  for (int i = 0; i < 16; ++i) {
    int k = i + 8 * half + (i & 8);
    a[i] = base[row * lda + k];
  }
  return a;
}
// B tile pre-swizzled in LDS: 512 f16 per 32x16 tile, ordered [lane][i]
// -> one aligned 32B load per lane (ds_load_b128 x2)
__device__ __forceinline__ v16h ldsB_swz(const f16* tileBase, int lane) {
  return *(const v16h*)(tileBase + lane * 16);
}

// swizzle writer: dst tile layout [lane*16 + i] <- W[(kt*32 + i + 8*half + (i&8)) * N + nt*16 + col]
__device__ __forceinline__ void write_swz_layer(
    int tid, int nthreads, const float* W, const float* g, const float* rv,
    int Kin, int Kpad, int N, f16* dst)
{
  int ntiles = N >> 4;
  int total  = Kpad * N;
  for (int s = tid; s < total; s += nthreads) {
    int tile   = s >> 9;
    int within = s & 511;
    int ln  = within >> 4;
    int i   = within & 15;
    int hf  = ln >> 4;
    int col = ln & 15;
    int kt  = tile / ntiles;
    int nt  = tile - kt * ntiles;
    int k   = kt * 32 + i + 8 * hf + (i & 8);
    int n   = nt * 16 + col;
    float sc = g[n] * rsqrtf(rv[n] + BN_EPS);
    float v  = (k < Kin) ? W[k * N + n] * sc : 0.f;
    dst[s] = (f16)v;
  }
}

// ---------------- Kernel 0: fold BN into f16 weights (B-swizzled) + f32 bias ----------------
__global__ void prep_weights(
    const float* W0, const float* b0, const float* g0, const float* be0, const float* rm0, const float* rv0,
    const float* W1, const float* b1, const float* g1, const float* be1, const float* rm1, const float* rv1,
    const float* W2, const float* b2, const float* g2, const float* be2, const float* rm2, const float* rv2,
    f16* __restrict__ Wh, float* __restrict__ biasOut)
{
  int tid = threadIdx.x;
  for (int o = tid; o < 64; o += 256) {
    float s = g0[o] * rsqrtf(rv0[o] + BN_EPS);
    biasOut[o] = (b0[o] - rm0[o]) * s + be0[o];
  }
  for (int o = tid; o < 64; o += 256) {
    float s = g1[o] * rsqrtf(rv1[o] + BN_EPS);
    biasOut[64 + o] = (b1[o] - rm1[o]) * s + be1[o];
  }
  for (int o = tid; o < 128; o += 256) {
    float s = g2[o] * rsqrtf(rv2[o] + BN_EPS);
    biasOut[128 + o] = (b2[o] - rm2[o]) * s + be2[o];
  }
  write_swz_layer(tid, 256, W0, g0, rv0, 67, 96, 64,  Wh);
  write_swz_layer(tid, 256, W1, g1, rv1, 64, 64, 64,  Wh + W0_ELEMS);
  write_swz_layer(tid, 256, W2, g2, rv2, 64, 64, 128, Wh + W0_ELEMS + W1_ELEMS);
}

// ---------------- Kernel 1: farthest point sampling (1 block / batch) ----------------
__global__ void __launch_bounds__(256) fps_kernel(
    const float* __restrict__ pos, int* __restrict__ fpsIdx, float* __restrict__ sampledOut)
{
  __shared__ float spos[NPTS * 3];
  __shared__ float rv_[256];
  __shared__ int   ri_[256];
  __shared__ int   curSh;
  int b = blockIdx.x, tid = threadIdx.x;
  for (int i = tid; i < NPTS * 3; i += 256) spos[i] = pos[b * NPTS * 3 + i];
  if (tid == 0) curSh = 0;
  float dist[16];
#pragma unroll
  for (int j = 0; j < 16; ++j) dist[j] = 1e10f;
  __syncthreads();

  for (int it = 0; it < NSAMP; ++it) {
    int cur = curSh;
    float cx = spos[cur * 3 + 0], cy = spos[cur * 3 + 1], cz = spos[cur * 3 + 2];
    if (tid == 0) {
      fpsIdx[b * NSAMP + it] = cur;
      sampledOut[(b * NSAMP + it) * 3 + 0] = cx;
      sampledOut[(b * NSAMP + it) * 3 + 1] = cy;
      sampledOut[(b * NSAMP + it) * 3 + 2] = cz;
    }
    float bestV = -1.f; int bestI = 0;
#pragma unroll
    for (int j = 0; j < 16; ++j) {
      int p = tid + 256 * j;
      float dx = spos[p * 3 + 0] - cx, dy = spos[p * 3 + 1] - cy, dz = spos[p * 3 + 2] - cz;
      float d = dx * dx + dy * dy + dz * dz;
      dist[j] = fminf(dist[j], d);
      if (dist[j] > bestV) { bestV = dist[j]; bestI = p; }
    }
    rv_[tid] = bestV; ri_[tid] = bestI;
    __syncthreads();
    for (int s = 128; s > 0; s >>= 1) {
      if (tid < s) {
        float v2 = rv_[tid + s]; int i2 = ri_[tid + s];
        if (v2 > rv_[tid] || (v2 == rv_[tid] && i2 < ri_[tid])) { rv_[tid] = v2; ri_[tid] = i2; }
      }
      __syncthreads();
    }
    if (tid == 0) curSh = ri_[0];
    __syncthreads();
  }
}

// ---------------- Kernel 2: kNN top-64 per centroid (1 block / centroid) ----------------
__global__ void __launch_bounds__(256) topk_kernel(
    const float* __restrict__ pos, const float* __restrict__ sampled,
    int* __restrict__ tkIdx, float* __restrict__ tkDist)
{
  __shared__ float sd[NPTS];
  __shared__ float rv_[256];
  __shared__ int   ri_[256];
  int cen = blockIdx.x;          // b*NSAMP + s
  int b = cen >> 10;
  int tid = threadIdx.x;
  float sx = sampled[cen * 3 + 0], sy = sampled[cen * 3 + 1], sz = sampled[cen * 3 + 2];
#pragma unroll
  for (int j = 0; j < 16; ++j) {
    int p = tid + 256 * j;
    const float* pp = &pos[(b * NPTS + p) * 3];
    float dx = pp[0] - sx, dy = pp[1] - sy, dz = pp[2] - sz;
    sd[p] = sqrtf(fmaxf(dx * dx + dy * dy + dz * dz, 0.f));
  }
  __syncthreads();
  for (int r = 0; r < KNN; ++r) {
    float bestV = 3.0e38f; int bestI = 0;
#pragma unroll
    for (int j = 0; j < 16; ++j) {
      int p = tid + 256 * j;
      float d = sd[p];
      if (d < bestV) { bestV = d; bestI = p; }
    }
    rv_[tid] = bestV; ri_[tid] = bestI;
    __syncthreads();
    for (int s = 128; s > 0; s >>= 1) {
      if (tid < s) {
        float v2 = rv_[tid + s]; int i2 = ri_[tid + s];
        if (v2 < rv_[tid] || (v2 == rv_[tid] && i2 < ri_[tid])) { rv_[tid] = v2; ri_[tid] = i2; }
      }
      __syncthreads();
    }
    if (tid == 0) {
      tkIdx[cen * KNN + r]  = ri_[0];
      tkDist[cen * KNN + r] = rv_[0];
      sd[ri_[0]] = 3.0e38f;
    }
    __syncthreads();
  }
}

// ---------------- Kernel 3: fused gather + 3x(GEMM+BN+ReLU) + masked max-pool ----------------
// 1 block (4 waves) per centroid. M=64 neighbors, WMMA 16x16x32 f16->f32.
__global__ void __launch_bounds__(128) mlp_kernel(
    const float* __restrict__ x, const float* __restrict__ pos,
    const float* __restrict__ sampled, const int* __restrict__ tkIdx,
    const float* __restrict__ tkDist, const f16* __restrict__ Wh,
    const float* __restrict__ biasWs, float* __restrict__ out)
{
  __shared__ __align__(32) f16   sA[64 * KP0];    // layer-0 features; reused as H2 (64x64)
  __shared__ __align__(32) f16   sH1[64 * 64];
  __shared__ __align__(32) f16   sWbuf[8192];     // per-layer B-swizzled weights
  __shared__ float sBias[256];
  __shared__ int   sNbr[KNN];
  __shared__ float sDist[KNN];
  __shared__ float sPart[4 * 128];

  int cen  = blockIdx.x;
  int b    = cen >> 10;
  int tid  = threadIdx.x;
  int lane = tid & 31;
  int w    = tid >> 5;
  int half = lane >> 4;
  int l16  = lane & 15;
  int mbase = w * 16;

  if (tid < KNN) {
    sNbr[tid]  = tkIdx[cen * KNN + tid];
    sDist[tid] = tkDist[cen * KNN + tid];
  }
  for (int i = tid; i < 256; i += 128) sBias[i] = biasWs[i];
  for (int i = tid; i < W0_ELEMS; i += 128) sWbuf[i] = Wh[i];
  __syncthreads();

  // gather grouped features: [rel_pos(3) | x(64) | pad(29)] -> f16 64x96
  float sx = sampled[cen * 3 + 0], sy = sampled[cen * 3 + 1], sz = sampled[cen * 3 + 2];
  for (int i = tid; i < 64 * KP0; i += 128) {
    int row = i / KP0, c = i - row * KP0;
    int n = sNbr[row];
    float v;
    if (c < 3) {
      float pv = pos[(b * NPTS + n) * 3 + c];
      v = pv - ((c == 0) ? sx : (c == 1) ? sy : sz);
    } else if (c < 67) {
      v = x[(b * NPTS + n) * CFEAT + (c - 3)];
    } else {
      v = 0.f;
    }
    sA[i] = (f16)v;
  }
  __syncthreads();

  // ---- Layer 0: (64x96) x (96x64), A fragments hoisted ----
  {
    v16h a0 = ldsA_frag(&sA[mbase * KP0 +  0], KP0, lane);
    v16h a1 = ldsA_frag(&sA[mbase * KP0 + 32], KP0, lane);
    v16h a2 = ldsA_frag(&sA[mbase * KP0 + 64], KP0, lane);
#pragma unroll
    for (int nt = 0; nt < 4; ++nt) {
      v8f acc = {};
      acc = __builtin_amdgcn_wmma_f32_16x16x32_f16(false, a0, false, ldsB_swz(&sWbuf[(0 * 4 + nt) * 512], lane), (short)0, acc, false, false);
      acc = __builtin_amdgcn_wmma_f32_16x16x32_f16(false, a1, false, ldsB_swz(&sWbuf[(1 * 4 + nt) * 512], lane), (short)0, acc, false, false);
      acc = __builtin_amdgcn_wmma_f32_16x16x32_f16(false, a2, false, ldsB_swz(&sWbuf[(2 * 4 + nt) * 512], lane), (short)0, acc, false, false);
#pragma unroll
      for (int r = 0; r < 8; ++r) {
        int row = mbase + r + 8 * half;
        int col = nt * 16 + l16;
        sH1[row * 64 + col] = (f16)fmaxf(acc[r] + sBias[col], 0.f);
      }
    }
  }
  __syncthreads();
  for (int i = tid; i < W1_ELEMS; i += 128) sWbuf[i] = Wh[W0_ELEMS + i];
  __syncthreads();

  // ---- Layer 1: (64x64) x (64x64) ----  (H2 overwrites sA)
  f16* sH2 = sA;
  {
    v16h a0 = ldsA_frag(&sH1[mbase * 64 +  0], 64, lane);
    v16h a1 = ldsA_frag(&sH1[mbase * 64 + 32], 64, lane);
#pragma unroll
    for (int nt = 0; nt < 4; ++nt) {
      v8f acc = {};
      acc = __builtin_amdgcn_wmma_f32_16x16x32_f16(false, a0, false, ldsB_swz(&sWbuf[(0 * 4 + nt) * 512], lane), (short)0, acc, false, false);
      acc = __builtin_amdgcn_wmma_f32_16x16x32_f16(false, a1, false, ldsB_swz(&sWbuf[(1 * 4 + nt) * 512], lane), (short)0, acc, false, false);
#pragma unroll
      for (int r = 0; r < 8; ++r) {
        int row = mbase + r + 8 * half;
        int col = nt * 16 + l16;
        sH2[row * 64 + col] = (f16)fmaxf(acc[r] + sBias[64 + col], 0.f);
      }
    }
  }
  __syncthreads();
  for (int i = tid; i < W2_ELEMS; i += 128) sWbuf[i] = Wh[W0_ELEMS + W1_ELEMS + i];
  __syncthreads();

  // ---- Layer 2: (64x64) x (64x128) + radius-masked max pool over rows ----
  {
    v16h a0 = ldsA_frag(&sH2[mbase * 64 +  0], 64, lane);
    v16h a1 = ldsA_frag(&sH2[mbase * 64 + 32], 64, lane);
#pragma unroll
    for (int nt = 0; nt < 8; ++nt) {
      v8f acc = {};
      acc = __builtin_amdgcn_wmma_f32_16x16x32_f16(false, a0, false, ldsB_swz(&sWbuf[(0 * 8 + nt) * 512], lane), (short)0, acc, false, false);
      acc = __builtin_amdgcn_wmma_f32_16x16x32_f16(false, a1, false, ldsB_swz(&sWbuf[(1 * 8 + nt) * 512], lane), (short)0, acc, false, false);
      int col = nt * 16 + l16;
      float m = NEGV;
#pragma unroll
      for (int r = 0; r < 8; ++r) {
        int row = mbase + r + 8 * half;            // row == neighbor k
        float v = fmaxf(acc[r] + sBias[128 + col], 0.f);
        v = (sDist[row] <= RADIUS) ? v : NEGV;
        m = fmaxf(m, v);
      }
      float o = __shfl_xor(m, 16, 32);             // merge the two half-wave row groups
      m = fmaxf(m, o);
      if (half == 0) sPart[w * 128 + col] = m;
    }
  }
  __syncthreads();
  {
    float m0 = fmaxf(sPart[tid],       sPart[128 + tid]);
    float m1 = fmaxf(sPart[256 + tid], sPart[384 + tid]);
    out[cen * 128 + tid] = fmaxf(m0, m1);
  }
}

// ---------------- host launch ----------------
extern "C" void kernel_launch(void* const* d_in, const int* in_sizes, int n_in,
                              void* d_out, int out_size, void* d_ws, size_t ws_size,
                              hipStream_t stream) {
  const float* x   = (const float*)d_in[0];
  const float* pos = (const float*)d_in[1];
  const float* W0  = (const float*)d_in[2];
  const float* b0  = (const float*)d_in[3];
  const float* g0  = (const float*)d_in[4];
  const float* be0 = (const float*)d_in[5];
  const float* rm0 = (const float*)d_in[6];
  const float* rv0 = (const float*)d_in[7];
  const float* W1  = (const float*)d_in[8];
  const float* b1  = (const float*)d_in[9];
  const float* g1  = (const float*)d_in[10];
  const float* be1 = (const float*)d_in[11];
  const float* rm1 = (const float*)d_in[12];
  const float* rv1 = (const float*)d_in[13];
  const float* W2  = (const float*)d_in[14];
  const float* b2  = (const float*)d_in[15];
  const float* g2  = (const float*)d_in[16];
  const float* be2 = (const float*)d_in[17];
  const float* rm2 = (const float*)d_in[18];
  const float* rv2 = (const float*)d_in[19];

  char* ws       = (char*)d_ws;
  f16*   Wh      = (f16*)(ws + WH_OFF);
  float* biasWs  = (float*)(ws + BIAS_OFF);
  int*   fpsIdx  = (int*)(ws + FPS_OFF);
  int*   tkIdx   = (int*)(ws + TKI_OFF);
  float* tkDist  = (float*)(ws + TKD_OFF);

  float* out        = (float*)d_out;                 // (B,NS,128)
  float* sampledOut = out + BZ * NSAMP * 128;        // (B,NS,3) appended

  prep_weights<<<1, 256, 0, stream>>>(W0, b0, g0, be0, rm0, rv0,
                                      W1, b1, g1, be1, rm1, rv1,
                                      W2, b2, g2, be2, rm2, rv2,
                                      Wh, biasWs);
  fps_kernel<<<BZ, 256, 0, stream>>>(pos, fpsIdx, sampledOut);
  topk_kernel<<<BZ * NSAMP, 256, 0, stream>>>(pos, sampledOut, tkIdx, tkDist);
  mlp_kernel<<<BZ * NSAMP, 128, 0, stream>>>(x, pos, sampledOut, tkIdx, tkDist, Wh, biasWs, out);
}
